// AttentionMambaBlock_30159260352666
// MI455X (gfx1250) — matrix operand order repaired and checked by reading, not compiled, not run on hardware
//
#include <hip/hip_runtime.h>
#include <hip/hip_bf16.h>
#include <math.h>

// ---------------------------------------------------------------------------
// Types for CDNA5 WMMA (wave32): D(f32 16x16) = A(bf16 16x32) x B(bf16 32x16) + C
// ---------------------------------------------------------------------------
typedef __attribute__((ext_vector_type(16))) __bf16 v16bf;
typedef __attribute__((ext_vector_type(8)))  float  v8f;
typedef __attribute__((ext_vector_type(4)))  unsigned v4u;
typedef __attribute__((ext_vector_type(8)))  int      v8i;
typedef __attribute__((ext_vector_type(4)))  int      v4i;

#define WMMA_BF16(A, B, C) \
  __builtin_amdgcn_wmma_f32_16x16x32_bf16(false, (A), false, (B), (short)0, (C), false, false)

static __device__ __forceinline__ float sigmoidf_(float x) {
  return 1.0f / (1.0f + __expf(-x));
}

// LDS byte offset of a shared-memory pointer: generic (flat) LDS addresses map
// to the LDS aperture with the LDS byte address in addr[31:0] (ISA 10.2).
static __device__ __forceinline__ unsigned lds_off_(const void* p) {
  return (unsigned)(unsigned long long)p;
}

// Async global->LDS copy of 16 bytes per lane (GLOBAL_LOAD_ASYNC_TO_LDS_B128).
// Same INST_OFFSET applies to both global and LDS addresses (ISA 08 §4.4).
static __device__ __forceinline__ void async_g2l_b128(unsigned lds, const float* g) {
  asm volatile("global_load_async_to_lds_b128 %0, %1, off" :: "v"(lds), "v"(g) : "memory");
}
static __device__ __forceinline__ void async_g2l_b128_o16(unsigned lds, const float* g) {
  asm volatile("global_load_async_to_lds_b128 %0, %1, off offset:16" :: "v"(lds), "v"(g) : "memory");
}
static __device__ __forceinline__ void wait_async0_() {
  asm volatile("s_wait_asynccnt 0x0" ::: "memory");
}

// ---------------------------------------------------------------------------
// Tensor Data Mover: load a dense 2D tile (tile_d0 x tile_d1 elements of 4B)
// from a row-major tensor (row stride stride0 elements) into LDS at lds_addr.
// Rows/cols beyond tensor_d0/tensor_d1 are zero-filled by the TDM (OOB reads
// return zero, ISA 08 §8.2) -- used to handle the N-boundary weight tile.
// D# packing per CDNA5 ISA.
// This toolchain (clang-23/therock-10) uses the 6-arg builtin form:
//   (uint32x4 g0, int32x8 g1, int32x4 g2, int32x4 g3, int32x8 g4, i32 cpol)
// ---------------------------------------------------------------------------
static __device__ __forceinline__ void tdm_load_2d(const void* gptr, unsigned lds_addr,
                                                   unsigned tensor_d0, unsigned tensor_d1,
                                                   unsigned tile_d0, unsigned tile_d1,
                                                   unsigned stride0_elems)
{
  const unsigned long long ga = (unsigned long long)gptr;
  v4u g0;
  g0[0] = 1u;                                   // count=1 (valid), user mode, no gather
  g0[1] = lds_addr;                             // lds_addr [63:32]
  g0[2] = (unsigned)ga;                         // global_addr [95:64]
  g0[3] = (unsigned)(ga >> 32) | (2u << 30);    // global_addr [120:96] | type=2 ("image")
  v8i g1;
  g1[0] = (int)(2u << 16);                      // wg_mask=0 | data_size=2 (4B) | no pad/iter
  g1[1] = (int)((tensor_d0 & 0xFFFFu) << 16);   // tensor_dim0[15:0] @ [63:48]
  g1[2] = (int)((tensor_d0 >> 16) | ((tensor_d1 & 0xFFFFu) << 16));  // d0[31:16] | d1[15:0]
  g1[3] = (int)((tensor_d1 >> 16) | (tile_d0 << 16));                // d1[31:16] | tile_dim0
  g1[4] = (int)(tile_d1 & 0xFFFFu);             // tile_dim1 | tile_dim2=0
  g1[5] = (int)stride0_elems;                   // tensor_dim0_stride[31:0]
  g1[6] = 0;                                    // stride0[47:32] | dim1_stride[15:0]
  g1[7] = 0;                                    // dim1_stride[47:16]
  v4i g2 = {0, 0, 0, 0};                        // 2D tensor: groups 2/3 unused
  v4i g3 = {0, 0, 0, 0};
  v8i g4 = {0, 0, 0, 0, 0, 0, 0, 0};            // extra group (6-arg toolchain form)
  __builtin_amdgcn_tensor_load_to_lds(g0, g1, g2, g3, g4, 0);
}

// ---------------------------------------------------------------------------
// Generic WMMA GEMM:  Y[M,N] = act( X[M,K] @ W[N,K]^T + bias[N] + resid[M,N] )
//   - 64x64 output tile / block, 256 threads (8 wave32 waves)
//   - each wave: 32(M) x 16(N) via two 16x16 accumulators
//   - K stepped by 32; X tile staged via per-lane async-to-LDS copies,
//     W tile staged by one Tensor Data Mover descriptor (zero-fills N edge);
//     fp32 tiles converted to bf16 while building WMMA fragments
//   - act: 0 = none, 1 = softplus
// ---------------------------------------------------------------------------
#define BM 64
#define BN 64
#define BK 32

__global__ __launch_bounds__(256)
void wmma_gemm_kernel(const float* __restrict__ X, int lda,
                      const float* __restrict__ W, int ldw,
                      const float* __restrict__ bias,
                      const float* __restrict__ resid, int ldr,
                      float* __restrict__ Y, int ldy,
                      int M, int N, int K, int act)
{
  __shared__ float As32[BM][BK];
  __shared__ float Bs32[BN][BK];

  const int tid   = threadIdx.x;
  const int wave  = tid >> 5;
  const int lane  = tid & 31;
  const int lhalf = lane >> 4;     // 0: lanes 0-15, 1: lanes 16-31
  const int lm    = lane & 15;
  const int wm    = wave & 1;      // M 32-block within tile
  const int wn    = wave >> 1;     // N 16-block within tile (0..3)
  const int mBase = blockIdx.y * BM;
  const int nBase = blockIdx.x * BN;

  const int ldRow = tid >> 2;        // 0..63
  const int ldCol = (tid & 3) * 8;   // 0,8,16,24

  const unsigned ldsA  = lds_off_(&As32[ldRow][ldCol]);
  const unsigned ldsB0 = lds_off_(&Bs32[0][0]);
  const unsigned wRows = (unsigned)((N - nBase) < BN ? (N - nBase) : BN);

  v8f c0 = {};
  v8f c1 = {};

  for (int k0 = 0; k0 < K; k0 += BK) {
    // ---- stage W tile (64 x 32 fp32) with one TDM descriptor (wave 0) ----
    if (wave == 0) {
      tdm_load_2d(W + (size_t)nBase * ldw + k0, ldsB0,
                  /*tensor_d0=*/BK, /*tensor_d1=*/wRows,
                  /*tile_d0=*/BK, /*tile_d1=*/BN, /*stride0=*/(unsigned)ldw);
    }
    // ---- stage X tile (64 x 32 fp32) via async copy engine (all waves) ----
    {
      const float* src = X + (size_t)(mBase + ldRow) * lda + k0 + ldCol;
      if (k0 + BK < K) __builtin_prefetch(src + BK, 0, 1);  // global_prefetch_b8
      async_g2l_b128(ldsA, src);
      async_g2l_b128_o16(ldsA, src);
    }
    wait_async0_();
    if (wave == 0) __builtin_amdgcn_s_wait_tensorcnt(0);
    __syncthreads();

    // ---- build bf16 fragments per documented 16-bit lane layouts ----
    v16bf a0, a1, bfrag;
    #pragma unroll
    for (int e = 0; e < 16; ++e) {
      const int kk = ((e >> 3) << 4) + (lhalf << 3) + (e & 7);  // A: k = 16*(e/8)+8*half+(e%8)
      a0[e]    = (__bf16)As32[(wm << 5) + lm][kk];
      a1[e]    = (__bf16)As32[(wm << 5) + 16 + lm][kk];
      bfrag[e] = (__bf16)Bs32[(wn << 4) + lm][kk]; // B: k = 16*(e/8)+8*half+(e%8)
    }
    c0 = WMMA_BF16(a0, bfrag, c0);
    c1 = WMMA_BF16(a1, bfrag, c1);
    __syncthreads();
  }

  // ---- epilogue (C layout: elem j -> row = 8*half + j, col = lane&15) ----
  const int col = nBase + (wn << 4) + lm;
  if (col < N) {
    const float bv = bias ? bias[col] : 0.0f;
    #pragma unroll
    for (int j = 0; j < 8; ++j) {
      const int r0 = mBase + (wm << 5) + (lhalf << 3) + j;
      const int r1 = r0 + 16;
      float v0c = c0[j] + bv;
      float v1c = c1[j] + bv;
      if (resid) {
        v0c += resid[(size_t)r0 * ldr + col];
        v1c += resid[(size_t)r1 * ldr + col];
      }
      if (act == 1) {  // softplus
        v0c = (v0c > 20.0f) ? v0c : log1pf(__expf(v0c));
        v1c = (v1c > 20.0f) ? v1c : log1pf(__expf(v1c));
      }
      if (r0 < M) Y[(size_t)r0 * ldy + col] = v0c;
      if (r1 < M) Y[(size_t)r1 * ldy + col] = v1c;
    }
  }
}

// ---------------------------------------------------------------------------
// Flash attention: one wave32 per (batch, head, 16-query block).
// Layout of Q/K/V/ctx buffers: [B*L, H] with in-row index = head*64 + d.
// Online softmax; scores and P@V through WMMA; P re-shaped C->A layout via LDS.
// ---------------------------------------------------------------------------
#define FAW 4   // waves per block

__global__ __launch_bounds__(128)
void flash_attn_kernel(const float* __restrict__ Q, const float* __restrict__ Kp,
                       const float* __restrict__ Vp, const int* __restrict__ mask,
                       float* __restrict__ Yc)
{
  __shared__ __bf16 Plds[FAW][16][32];

  const int wave  = threadIdx.x >> 5;
  const int lane  = threadIdx.x & 31;
  const int lhalf = lane >> 4;
  const int lm    = lane & 15;

  const int unit = blockIdx.x * FAW + wave;   // B*NH*(L/16) = 2048 units
  const int qblk = unit & 63;
  const int h    = (unit >> 6) & 15;
  const int b    = unit >> 10;

  // ---- load Q fragments once (A layout; d-chunks 0-31 and 32-63) ----
  const int qrow = qblk * 16 + lm;
  const float* qp = Q + (size_t)(b * 1024 + qrow) * 1024 + h * 64;
  v16bf aq0, aq1;
  #pragma unroll
  for (int g = 0; g < 2; ++g) {
    const float* p0 = qp + lhalf * 16 + g * 8;        // chunk 0: d in [0,32)
    const float* p1 = qp + 32 + lhalf * 16 + g * 8;   // chunk 1: d in [32,64)
    #pragma unroll
    for (int i = 0; i < 8; ++i) {
      aq0[g * 8 + i] = (__bf16)p0[i];
      aq1[g * 8 + i] = (__bf16)p1[i];
    }
  }

  float rmax[8], rsum[8];
  #pragma unroll
  for (int j = 0; j < 8; ++j) { rmax[j] = -3.0e38f; rsum[j] = 0.0f; }
  v8f o[4] = {};

  for (int kv = 0; kv < 1024; kv += 32) {
    // ---- S = Q K^T for 32 keys (two 16-key tiles) ----
    v8f s0 = {}, s1 = {};
    #pragma unroll
    for (int t = 0; t < 2; ++t) {
      const int key = kv + t * 16 + lm;                      // B-frag: n = key
      const float* kp = Kp + (size_t)(b * 1024 + key) * 1024 + h * 64;
      v16bf bk0, bk1;
      const float* p0 = kp + lhalf * 16;                     // k = 16*half + e, d-chunk 0
      const float* p1 = kp + 32 + lhalf * 16;                // d-chunk 1
      #pragma unroll
      for (int e = 0; e < 16; ++e) {
        bk0[e] = (__bf16)p0[e];
        bk1[e] = (__bf16)p1[e];
      }
      if (t == 0) { s0 = WMMA_BF16(aq0, bk0, s0); s0 = WMMA_BF16(aq1, bk1, s0); }
      else        { s1 = WMMA_BF16(aq0, bk0, s1); s1 = WMMA_BF16(aq1, bk1, s1); }
    }

    const float mk0 = (float)mask[b * 1024 + kv + lm];
    const float mk1 = (float)mask[b * 1024 + kv + 16 + lm];

    // ---- online softmax; lanes 0-15 / 16-31 independently hold rows j / j+8 ----
    #pragma unroll
    for (int j = 0; j < 8; ++j) {
      float v0c = s0[j] * 0.125f + mk0;   // 1/sqrt(64)
      float v1c = s1[j] * 0.125f + mk1;
      float mx = fmaxf(v0c, v1c);
      for (int off = 1; off < 16; off <<= 1) mx = fmaxf(mx, __shfl_xor(mx, off, 32));
      const float nm  = fmaxf(rmax[j], mx);
      const float p0v = __expf(v0c - nm);
      const float p1v = __expf(v1c - nm);
      float ls = p0v + p1v;
      for (int off = 1; off < 16; off <<= 1) ls += __shfl_xor(ls, off, 32);
      const float al = __expf(rmax[j] - nm);
      rsum[j] = rsum[j] * al + ls;
      rmax[j] = nm;
      o[0][j] *= al; o[1][j] *= al; o[2][j] *= al; o[3][j] *= al;
      // stash P in LDS (C layout: row = 8*half + j, cols lm / 16+lm)
      Plds[wave][(lhalf << 3) + j][lm]      = (__bf16)p0v;
      Plds[wave][(lhalf << 3) + j][16 + lm] = (__bf16)p1v;
    }
    asm volatile("s_wait_dscnt 0x0" ::: "memory");

    // ---- reload P as A-fragment (16 q rows x 32 keys) ----
    v16bf ap;
    #pragma unroll
    for (int e = 0; e < 16; ++e) {
      const int kk = (lhalf << 4) + e;
      ap[e] = Plds[wave][lm][kk];
    }

    // ---- O += P @ V  (four 16-wide d chunks) ----
    #pragma unroll
    for (int cc = 0; cc < 4; ++cc) {
      v16bf bvf;
      #pragma unroll
      for (int e = 0; e < 16; ++e) {
        const int key = kv + (lhalf << 4) + e;   // B-frag: k = key, n = d
        bvf[e] = (__bf16)Vp[(size_t)(b * 1024 + key) * 1024 + h * 64 + cc * 16 + lm];
      }
      o[cc] = WMMA_BF16(ap, bvf, o[cc]);
    }
  }

  // ---- normalize and store ctx ----
  #pragma unroll
  for (int cc = 0; cc < 4; ++cc) {
    #pragma unroll
    for (int j = 0; j < 8; ++j) {
      const int r = qblk * 16 + (lhalf << 3) + j;
      Yc[(size_t)(b * 1024 + r) * 1024 + h * 64 + cc * 16 + lm] = o[cc][j] / rsum[j];
    }
  }
}

// ---------------------------------------------------------------------------
// Fused: attn_out = LayerNorm(t0) ; x2 = attn_out + x ; h = RMSNorm(x2)*w
// t0 already contains ctx@Wo^T + bo + x (GEMM epilogue). One block per row.
// ---------------------------------------------------------------------------
__global__ __launch_bounds__(256)
void ln_resid_kernel(const float* __restrict__ t0, const float* __restrict__ x,
                     const float* __restrict__ lnw, const float* __restrict__ lnb,
                     const float* __restrict__ rmsw,
                     float* __restrict__ x2, float* __restrict__ hout)
{
  __shared__ float red[256];
  const int row = blockIdx.x;
  const int tid = threadIdx.x;
  const float* u  = t0 + (size_t)row * 1024;
  const float* xr = x  + (size_t)row * 1024;

  float uv[4], xv[4];
  float s = 0.0f;
  #pragma unroll
  for (int i = 0; i < 4; ++i) {
    uv[i] = u[tid + 256 * i];
    xv[i] = xr[tid + 256 * i];
    s += uv[i];
  }
  red[tid] = s; __syncthreads();
  for (int st = 128; st > 0; st >>= 1) { if (tid < st) red[tid] += red[tid + st]; __syncthreads(); }
  const float mu = red[0] * (1.0f / 1024.0f); __syncthreads();

  float s2 = 0.0f;
  #pragma unroll
  for (int i = 0; i < 4; ++i) { const float d = uv[i] - mu; s2 += d * d; }
  red[tid] = s2; __syncthreads();
  for (int st = 128; st > 0; st >>= 1) { if (tid < st) red[tid] += red[tid + st]; __syncthreads(); }
  const float inv = rsqrtf(red[0] * (1.0f / 1024.0f) + 1e-12f); __syncthreads();

  float x2v[4];
  float s3 = 0.0f;
  #pragma unroll
  for (int i = 0; i < 4; ++i) {
    const int c = tid + 256 * i;
    const float y = (uv[i] - mu) * inv * lnw[c] + lnb[c];
    x2v[i] = y + xv[i];
    s3 += x2v[i] * x2v[i];
  }
  red[tid] = s3; __syncthreads();
  for (int st = 128; st > 0; st >>= 1) { if (tid < st) red[tid] += red[tid + st]; __syncthreads(); }
  const float rinv = rsqrtf(red[0] * (1.0f / 1024.0f) + 1e-6f);

  #pragma unroll
  for (int i = 0; i < 4; ++i) {
    const int c = tid + 256 * i;
    x2[(size_t)row * 1024 + c]   = x2v[i];
    hout[(size_t)row * 1024 + c] = x2v[i] * rinv * rmsw[c];
  }
}

// ---------------------------------------------------------------------------
// Depthwise causal conv (K=4) + bias + SiLU + mask.  hs = proj[:, :2048]*mask.
// ---------------------------------------------------------------------------
__global__ __launch_bounds__(256)
void conv_silu_kernel(const float* __restrict__ proj, const float* __restrict__ cw,
                      const float* __restrict__ cb, const int* __restrict__ mask,
                      float* __restrict__ ssm_in)
{
  const int idx = blockIdx.x * blockDim.x + threadIdx.x;   // B*L*D_IN
  if (idx >= 2 * 1024 * 2048) return;
  const int c = idx & 2047;
  const int l = (idx >> 11) & 1023;
  const int b = idx >> 21;

  float acc = cb[c];
  #pragma unroll
  for (int j = 0; j < 4; ++j) {
    const int ls = l - 3 + j;
    if (ls >= 0) {
      const float hv = proj[(size_t)(b * 1024 + ls) * 4096 + c] * (float)mask[b * 1024 + ls];
      acc += hv * cw[c * 4 + j];
    }
  }
  const float sv = acc * sigmoidf_(acc) * (float)mask[b * 1024 + l];
  ssm_in[(size_t)(b * 1024 + l) * 2048 + c] = sv;
}

// ---------------------------------------------------------------------------
// Selective scan: one thread per (b, channel d); 16-state recurrence over L.
// Fused: y = scan + u*D ; out = y * silu(gate)   (gate = proj[:, 2048+d])
// ---------------------------------------------------------------------------
__global__ __launch_bounds__(256)
void scan_kernel(const float* __restrict__ dt, const float* __restrict__ u,
                 const float* __restrict__ dbc, const float* __restrict__ proj,
                 const float* __restrict__ A_log, const float* __restrict__ Dsk,
                 float* __restrict__ out)
{
  const int idx = blockIdx.x * blockDim.x + threadIdx.x;   // B*D_IN = 4096
  if (idx >= 2 * 2048) return;
  const int d = idx & 2047;
  const int b = idx >> 11;

  float Arow[16], st[16];
  #pragma unroll
  for (int n = 0; n < 16; ++n) {
    Arow[n] = -__expf(A_log[d * 16 + n]);
    st[n]   = 0.0f;
  }
  const float Dd = Dsk[d];

  for (int l = 0; l < 1024; ++l) {
    const size_t r  = (size_t)(b * 1024 + l);
    const float dtv = dt[r * 2048 + d];
    const float uv  = u[r * 2048 + d];
    const float* bc = dbc + r * 96;
    float y = 0.0f;
    #pragma unroll
    for (int n = 0; n < 16; ++n) {
      const float Bn = bc[64 + n];
      const float Cn = bc[80 + n];
      st[n] = __expf(dtv * Arow[n]) * st[n] + dtv * Bn * uv;
      y += st[n] * Cn;
    }
    const float g  = proj[r * 4096 + 2048 + d];
    out[r * 2048 + d] = (y + uv * Dd) * (g * sigmoidf_(g));
  }
}

// ---------------------------------------------------------------------------
// Final RMSNorm -> d_out
// ---------------------------------------------------------------------------
__global__ __launch_bounds__(256)
void rms_final_kernel(const float* __restrict__ x, const float* __restrict__ w,
                      float* __restrict__ out)
{
  __shared__ float red[256];
  const int row = blockIdx.x;
  const int tid = threadIdx.x;
  const float* xr = x + (size_t)row * 1024;
  float v[4];
  float s = 0.0f;
  #pragma unroll
  for (int i = 0; i < 4; ++i) { v[i] = xr[tid + 256 * i]; s += v[i] * v[i]; }
  red[tid] = s; __syncthreads();
  for (int st = 128; st > 0; st >>= 1) { if (tid < st) red[tid] += red[tid + st]; __syncthreads(); }
  const float inv = rsqrtf(red[0] * (1.0f / 1024.0f) + 1e-6f);
  #pragma unroll
  for (int i = 0; i < 4; ++i) {
    const int c = tid + 256 * i;
    out[(size_t)row * 1024 + c] = v[i] * inv * w[c];
  }
}

// ---------------------------------------------------------------------------
// Host launcher
// ---------------------------------------------------------------------------
extern "C" void kernel_launch(void* const* d_in, const int* in_sizes, int n_in,
                              void* d_out, int out_size, void* d_ws, size_t ws_size,
                              hipStream_t stream)
{
  (void)in_sizes; (void)n_in; (void)out_size; (void)ws_size;

  const float* hid  = (const float*)d_in[0];
  const int*   mask = (const int*)d_in[1];
  const float* Wq = (const float*)d_in[2];   const float* bq = (const float*)d_in[3];
  const float* Wk = (const float*)d_in[4];   const float* bk = (const float*)d_in[5];
  const float* Wv = (const float*)d_in[6];   const float* bv = (const float*)d_in[7];
  const float* Wo = (const float*)d_in[8];   const float* bo = (const float*)d_in[9];
  const float* lnw = (const float*)d_in[10]; const float* lnb = (const float*)d_in[11];
  const float* mnw = (const float*)d_in[12];
  const float* inw = (const float*)d_in[13];
  const float* cw  = (const float*)d_in[14]; const float* cb  = (const float*)d_in[15];
  const float* xpw = (const float*)d_in[16];
  const float* dtw = (const float*)d_in[17]; const float* dtbias = (const float*)d_in[18];
  const float* Alog = (const float*)d_in[19];
  const float* Dsk  = (const float*)d_in[20];
  const float* opw  = (const float*)d_in[21];
  const float* fnw  = (const float*)d_in[22];

  const size_t MT = 2048;  // B*L tokens
  float* ws  = (float*)d_ws;
  float* Qb  = ws; ws += MT * 1024;
  float* Kb  = ws; ws += MT * 1024;
  float* Vb  = ws; ws += MT * 1024;
  float* ctx = ws; ws += MT * 1024;
  float* t0  = ws; ws += MT * 1024;
  float* x2  = ws; ws += MT * 1024;
  float* hb  = ws; ws += MT * 1024;
  float* proj = ws; ws += MT * 4096;
  float* ssm  = ws; ws += MT * 2048;
  float* dbc  = ws; ws += MT * 96;
  float* dtv  = ws; ws += MT * 2048;
  float* sout = ws; ws += MT * 2048;
  float* x3   = ws; ws += MT * 1024;

  const dim3 blk(256);
  auto gemm = [&](const float* X, int lda, const float* W, int ldw, const float* bias,
                  const float* res, int ldr, float* Y, int ldy, int M, int N, int K, int act) {
    dim3 grid((N + BN - 1) / BN, (M + BM - 1) / BM);
    wmma_gemm_kernel<<<grid, blk, 0, stream>>>(X, lda, W, ldw, bias, res, ldr, Y, ldy, M, N, K, act);
  };

  // Attention
  gemm(hid, 1024, Wq, 1024, bq, nullptr, 0, Qb, 1024, 2048, 1024, 1024, 0);
  gemm(hid, 1024, Wk, 1024, bk, nullptr, 0, Kb, 1024, 2048, 1024, 1024, 0);
  gemm(hid, 1024, Wv, 1024, bv, nullptr, 0, Vb, 1024, 2048, 1024, 1024, 0);
  flash_attn_kernel<<<dim3(512), dim3(128), 0, stream>>>(Qb, Kb, Vb, mask, ctx);
  gemm(ctx, 1024, Wo, 1024, bo, hid, 1024, t0, 1024, 2048, 1024, 1024, 0);  // +bo +x
  ln_resid_kernel<<<dim3(2048), dim3(256), 0, stream>>>(t0, hid, lnw, lnb, mnw, x2, hb);

  // Mamba
  gemm(hb, 1024, inw, 1024, nullptr, nullptr, 0, proj, 4096, 2048, 4096, 1024, 0);
  conv_silu_kernel<<<dim3((2 * 1024 * 2048) / 256), dim3(256), 0, stream>>>(proj, cw, cb, mask, ssm);
  gemm(ssm, 2048, xpw, 2048, nullptr, nullptr, 0, dbc, 96, 2048, 96, 2048, 0);
  gemm(dbc, 96, dtw, 64, dtbias, nullptr, 0, dtv, 2048, 2048, 2048, 64, 1);  // softplus
  scan_kernel<<<dim3(16), dim3(256), 0, stream>>>(dtv, ssm, dbc, proj, Alog, Dsk, sout);
  gemm(sout, 2048, opw, 2048, nullptr, x2, 1024, x3, 1024, 2048, 1024, 2048, 0);  // + res2
  rms_final_kernel<<<dim3(2048), dim3(256), 0, stream>>>(x3, fnw, (float*)d_out);
}
